// Block_58394375356504
// MI455X (gfx1250) — matrix operand order repaired and checked by reading, not compiled, not run on hardware
//
#include <hip/hip_runtime.h>

// MI455X / gfx1250, wave32. Transformer block, all matmuls on
// v_wmma_f32_16x16x32_f16 (f16 in, f32 accumulate). GEMM A-tiles staged into
// LDS by the Tensor Data Mover (tensor_load_to_lds, TENSORcnt), B-tiles
// transposed manually for ds_load_b128 B-fragments.
// Fixed shape: B=2, T=2048, C=1024, H=16, DH=64.

#define USE_TDM 1

constexpr int kB  = 2;
constexpr int kT  = 2048;   // power of two -> shifts/masks, no runtime division
constexpr int kC  = 1024;
constexpr int kH  = 16;
constexpr int kDH = 64;

typedef __attribute__((ext_vector_type(16))) _Float16 v16h;
typedef __attribute__((ext_vector_type(8)))  _Float16 v8h;
typedef __attribute__((ext_vector_type(8)))  float    v8f;
typedef __attribute__((ext_vector_type(4)))  unsigned int u32x4;
typedef __attribute__((ext_vector_type(8)))  unsigned int u32x8;

static __device__ __forceinline__ v8f wmma32(v16h a, v16h b, v8f c) {
  // D = A(16x32 f16) * B(32x16 f16) + C(16x16 f32)
  return __builtin_amdgcn_wmma_f32_16x16x32_f16(false, a, false, b, (short)0, c,
                                                false, false);
}

// v16h from two v8h chunks (A-fragment halves, or contiguous B columns).
static __device__ __forceinline__ v16h ld_pair(const _Float16* p0,
                                               const _Float16* p1) {
  union { v16h v; v8h h[2]; } u;
  u.h[0] = *(const v8h*)p0;
  u.h[1] = *(const v8h*)p1;
  return u.v;
}

// ---------------------------------------------------------------------------
// Weight prep kernels
// ---------------------------------------------------------------------------
__global__ __launch_bounds__(256) void cvt_f32_f16(const float* __restrict__ in,
                                                   _Float16* __restrict__ out,
                                                   size_t n) {
  for (size_t i = (size_t)blockIdx.x * blockDim.x + threadIdx.x; i < n;
       i += (size_t)gridDim.x * blockDim.x)
    out[i] = (_Float16)in[i];
}

// Pack per-head W{q,k,v} [H,C,DH] f32 into one f16 matrix [C, 3*H*DH=3072],
// column = h*192 + which*64 + d  (so QKV is a single 4096x3072x1024 GEMM).
__global__ __launch_bounds__(256) void pack_qkv(const float* __restrict__ W,
                                                _Float16* __restrict__ out,
                                                int which) {
  const size_t n = (size_t)kH * kC * kDH;  // 1,048,576
  for (size_t i = (size_t)blockIdx.x * blockDim.x + threadIdx.x; i < n;
       i += (size_t)gridDim.x * blockDim.x) {
    int h = (int)(i >> 16);          // / (C*DH)
    int c = (int)((i >> 6) & 1023);  // / DH % C
    int d = (int)(i & 63);
    out[(size_t)c * (3 * kH * kDH) + h * 192 + which * 64 + d] =
        (_Float16)W[i];
  }
}

// ---------------------------------------------------------------------------
// LayerNorm over the TIME axis (dim 1), unbiased variance (ddof=1).
// ---------------------------------------------------------------------------
__global__ __launch_bounds__(256) void ln_time_kernel(
    const float* __restrict__ x, const float* __restrict__ g,
    const float* __restrict__ be, _Float16* __restrict__ out) {
  __shared__ float s1[8][32];
  __shared__ float s2[8][32];
  int tx = threadIdx.x & 31, ty = threadIdx.x >> 5;
  int b = blockIdx.x >> 5;              // groups = C/32 = 32
  int ch = (blockIdx.x & 31) * 32 + tx;
  const float* xp = x + (size_t)b * kT * kC + ch;

  float s = 0.f, ss = 0.f;
  for (int t = ty; t < kT; t += 8) {
    float v = xp[(size_t)t * kC];
    s += v;
    ss += v * v;
  }
  s1[ty][tx] = s;
  s2[ty][tx] = ss;
  __syncthreads();
  if (ty == 0) {
    float a = 0.f, bb = 0.f;
#pragma unroll
    for (int i = 0; i < 8; ++i) { a += s1[i][tx]; bb += s2[i][tx]; }
    float mean = a * (1.0f / kT);
    float var = (bb - (float)kT * mean * mean) * (1.0f / (kT - 1));
    s1[0][tx] = mean;
    s2[0][tx] = rsqrtf(var + 1e-5f);
  }
  __syncthreads();
  float mean = s1[0][tx], rstd = s2[0][tx];
  float gg = g[ch], bt = be[ch];
  _Float16* op = out + (size_t)b * kT * kC + ch;
  for (int t = ty; t < kT; t += 8) {
    float v = xp[(size_t)t * kC];
    op[(size_t)t * kC] = (_Float16)(gg * ((v - mean) * rstd) + bt);
  }
}

// ---------------------------------------------------------------------------
// WMMA GEMM, block tile 128x128, K step 32. 8 waves in a 4(m) x 2(n) grid;
// each wave owns a 32x64 patch = 2x4 WMMA tiles (8 v_wmma per K step).
//
// A tile (128x32 f16) is fetched by the Tensor Data Mover: wave 0 issues one
// tensor_load_to_lds per K step with a 2D D# (row stride = K elements) whose
// pad feature (4 DWORDs every 16 DWORDs) produces the 40-half padded LDS rows
// directly. Completion via s_wait_tensorcnt before the workgroup barrier.
// B tile staged TRANSPOSED (sBt[n][k]) so B-fragments are two ds_load_b128.
//
// Epilogue modes: 0 = f32 out (+bias,+f32 residual)
//                 1 = f16 out (+bias,+ReLU)
//                 2 = QKV: scatter to q[b,h,t,d], k[b,h,t,d], vT[b,h,d,t]
// ---------------------------------------------------------------------------
enum { EP_F32 = 0, EP_F16R = 1, EP_QKV = 2 };

template <int EP>
__global__ __launch_bounds__(256) void gemm128(
    const _Float16* __restrict__ A, const _Float16* __restrict__ Bw,
    const float* __restrict__ bias, const float* __restrict__ resid,
    void* __restrict__ o0, void* __restrict__ o1, void* __restrict__ o2,
    int N, int K) {
  __shared__ _Float16 sA[128][40];
  __shared__ _Float16 sBt[128][40];

  int tid = threadIdx.x;
  int lane = tid & 31, wid = tid >> 5;
  int lh = lane >> 4, ln = lane & 15;
  int wm = wid >> 1, wn = wid & 1;
  size_t m0 = (size_t)blockIdx.y * 128;
  int n0 = blockIdx.x * 128;

  v8f z = {0.f, 0.f, 0.f, 0.f, 0.f, 0.f, 0.f, 0.f};
  v8f acc[2][4];
#pragma unroll
  for (int ms = 0; ms < 2; ++ms)
#pragma unroll
    for (int ns = 0; ns < 4; ++ns) acc[ms][ns] = z;

  for (int k0 = 0; k0 < K; k0 += 32) {
    __syncthreads();  // previous tile fully consumed before overwrite

#if USE_TDM
    if (wid == 0) {
      // D# group 0: count=1 | lds_addr | 57-bit global addr | type=2
      unsigned int lds_off = (unsigned int)(uintptr_t)&sA[0][0];
      unsigned long long ga =
          (unsigned long long)(uintptr_t)&A[m0 * (size_t)K + k0];
      u32x4 g0 = {1u, lds_off, (unsigned int)ga,
                  (unsigned int)((ga >> 32) & 0x01FFFFFFu) | (2u << 30)};
      // D# group 1: data_size=2B; pad 4 DWORDs every 16 DWORDs (32 halves ->
      // 40-half rows); tensor_dim0=K, tensor_dim1=M=4096; tile 32x128;
      // tensor_dim0_stride=K.
      u32x8 g1 = {(1u << 16) | (1u << 20) | (3u << 22) | (3u << 25),
                  ((unsigned int)K & 0xFFFFu) << 16,
                  ((unsigned int)K >> 16) | (4096u << 16),
                  (32u << 16),
                  128u,
                  (unsigned int)K,
                  0u,
                  0u};
      asm volatile("tensor_load_to_lds %0, %1" ::"s"(g0), "s"(g1) : "memory");
    }
#else
#pragma unroll
    for (int it = 0; it < 2; ++it) {  // A tile: 128 rows x 32 halves
      int vi = tid + it * 256;
      int row = vi >> 2, cv = (vi & 3) * 8;
      *(v8h*)&sA[row][cv] = *(const v8h*)&A[(m0 + row) * (size_t)K + k0 + cv];
    }
#endif

    if (k0 + 32 < K)  // global_prefetch_b8 for the next B tile
      __builtin_prefetch(&Bw[(size_t)(k0 + 32 + (tid >> 4)) * N + n0], 0, 1);
#pragma unroll
    for (int it = 0; it < 2; ++it) {  // B tile: 32 k x 128 n, store transposed
      int vi = tid + it * 256;
      int kk = vi >> 4, ncv = (vi & 15) * 8;
      union { v8h v; _Float16 e[8]; } u;
      u.v = *(const v8h*)&Bw[(size_t)(k0 + kk) * N + n0 + ncv];
#pragma unroll
      for (int j = 0; j < 8; ++j) sBt[ncv + j][kk] = u.e[j];
    }

#if USE_TDM
    if (wid == 0) __builtin_amdgcn_s_wait_tensorcnt(0);
#endif
    __syncthreads();  // A (TDM) + B (ds stores) visible to all waves

    v16h af[2], bf[4];
#pragma unroll
    for (int ms = 0; ms < 2; ++ms) {
      const _Float16* p = &sA[wm * 32 + ms * 16 + ln][lh * 8];
      af[ms] = ld_pair(p, p + 16);
    }
#pragma unroll
    for (int ns = 0; ns < 4; ++ns) {
      const _Float16* p = &sBt[wn * 64 + ns * 16 + ln][lh * 8];
      bf[ns] = ld_pair(p, p + 16);  // K = 8h..8h+7, 16+8h..16+8h+7
    }
#pragma unroll
    for (int ms = 0; ms < 2; ++ms)
#pragma unroll
      for (int ns = 0; ns < 4; ++ns)
        acc[ms][ns] = wmma32(af[ms], bf[ns], acc[ms][ns]);
  }

  // Epilogue. C/D layout: VGPR r holds row (r + 8*laneHalf), col = lane&15.
#pragma unroll
  for (int ms = 0; ms < 2; ++ms)
#pragma unroll
    for (int ns = 0; ns < 4; ++ns)
#pragma unroll
      for (int r = 0; r < 8; ++r) {
        size_t row = m0 + wm * 32 + ms * 16 + r + 8 * lh;
        int col = n0 + wn * 64 + ns * 16 + ln;
        float v = acc[ms][ns][r];
        if (EP == EP_F32) {
          v += bias[col];
          if (resid) v += resid[row * N + col];
          ((float*)o0)[row * N + col] = v;
        } else if (EP == EP_F16R) {
          v += bias[col];
          v = fmaxf(v, 0.f);
          ((_Float16*)o0)[row * N + col] = (_Float16)v;
        } else {  // EP_QKV: col = h*192 + which*64 + d
          int b = (int)(row >> 11);        // / kT
          int t = (int)(row & (kT - 1));
          int h = col / 192;               // division by constant -> mul/shift
          int rem = col - h * 192;
          int which = rem >> 6;
          int d = rem & 63;
          size_t bh = (size_t)(b * kH + h);
          if (which == 0)
            ((_Float16*)o0)[(bh * kT + t) * 64 + d] = (_Float16)v;
          else if (which == 1)
            ((_Float16*)o1)[(bh * kT + t) * 64 + d] = (_Float16)v;
          else  // V stored transposed: vt[b,h,d,t]
            ((_Float16*)o2)[(bh * 64 + d) * kT + t] = (_Float16)v;
        }
      }
}

// ---------------------------------------------------------------------------
// Flash-style causal attention. 4 waves/WG, each wave owns 16 Q rows as
// A-fragments in registers. Per 32-key block: 4 WMMAs for S = Q K^T, online
// softmax (shfl_xor reductions over 16-lane row groups, __expf), P restriped
// through LDS into A-layout, 4 WMMAs accumulate P*V (V pre-transposed so each
// B-fragment is a single aligned 32B v16h load).
// ---------------------------------------------------------------------------
__global__ __launch_bounds__(128) void attn_kernel(
    const _Float16* __restrict__ q, const _Float16* __restrict__ k,
    const _Float16* __restrict__ vt, _Float16* __restrict__ attn,
    float scale) {
  __shared__ _Float16 sP[4][16][40];
  int tid = threadIdx.x;
  int lane = tid & 31, wave = tid >> 5;
  int lh = lane >> 4, ln = lane & 15;
  int b = blockIdx.z, h = blockIdx.y;
  size_t bh = (size_t)(b * kH + h);
  int m0 = blockIdx.x * 64 + wave * 16;
  int kend = blockIdx.x * 64 + 63;  // uniform per block -> __syncthreads safe

  const _Float16* qrow = q + (bh * kT + m0 + ln) * 64;
  v16h qf[2];
#pragma unroll
  for (int kc = 0; kc < 2; ++kc) {
    const _Float16* p = qrow + kc * 32 + lh * 16;
    qf[kc] = ld_pair(p, p + 8);
  }

  float mrow[8], lrow[8];
  v8f z = {0.f, 0.f, 0.f, 0.f, 0.f, 0.f, 0.f, 0.f};
  v8f accO[4] = {z, z, z, z};
#pragma unroll
  for (int r = 0; r < 8; ++r) { mrow[r] = -3.0e38f; lrow[r] = 0.f; }

  for (int kv = 0; kv <= kend; kv += 32) {
    v8f s[2] = {z, z};
#pragma unroll
    for (int n = 0; n < 2; ++n) {
      // K^T B-fragment: lane = key, 16 contiguous head dims per lane half.
      const _Float16* kp = k + (bh * kT + kv + n * 16 + ln) * 64 + lh * 16;
#pragma unroll
      for (int kc = 0; kc < 2; ++kc) {
        v16h bf = *(const v16h*)(kp + kc * 32);  // 32B-aligned contiguous
        s[n] = wmma32(qf[kc], bf, s[n]);
      }
    }
#pragma unroll
    for (int r = 0; r < 8; ++r) {
      int row = m0 + r + 8 * lh;
      float p0 = (float)s[0][r] * scale;
      if (kv + ln > row) p0 = -3.0e38f;
      float p1 = (float)s[1][r] * scale;
      if (kv + 16 + ln > row) p1 = -3.0e38f;
      float mx = fmaxf(p0, p1);
#pragma unroll
      for (int off = 1; off < 16; off <<= 1)
        mx = fmaxf(mx, __shfl_xor(mx, off, 32));
      float mn = fmaxf(mrow[r], mx);
      float e0 = __expf(p0 - mn), e1 = __expf(p1 - mn);
      float sm = e0 + e1;
#pragma unroll
      for (int off = 1; off < 16; off <<= 1) sm += __shfl_xor(sm, off, 32);
      float corr = __expf(mrow[r] - mn);
      lrow[r] = lrow[r] * corr + sm;
      mrow[r] = mn;
#pragma unroll
      for (int c = 0; c < 4; ++c) accO[c][r] *= corr;
      sP[wave][r + 8 * lh][ln] = (_Float16)e0;
      sP[wave][r + 8 * lh][16 + ln] = (_Float16)e1;
    }
    __syncthreads();
    const _Float16* pp = &sP[wave][ln][lh * 16];
    v16h pf = ld_pair(pp, pp + 8);  // P as A-fragment 16x32
#pragma unroll
    for (int c = 0; c < 4; ++c) {
      // V^T B-fragment: lane = head dim, 16 contiguous key positions.
      const _Float16* vp = vt + (bh * 64 + c * 16 + ln) * kT + kv + lh * 16;
      v16h vf = *(const v16h*)vp;  // single aligned 32B load
      accO[c] = wmma32(pf, vf, accO[c]);
    }
    __syncthreads();
  }

#pragma unroll
  for (int r = 0; r < 8; ++r) {
    float inv = 1.0f / lrow[r];
    int row = m0 + r + 8 * lh;
#pragma unroll
    for (int c = 0; c < 4; ++c)
      attn[((size_t)b * kT + row) * kC + h * 64 + c * 16 + ln] =
          (_Float16)((float)accO[c][r] * inv);
  }
}

// ---------------------------------------------------------------------------
extern "C" void kernel_launch(void* const* d_in, const int* in_sizes, int n_in,
                              void* d_out, int out_size, void* d_ws,
                              size_t ws_size, hipStream_t stream) {
  const size_t BT = (size_t)kB * kT;  // 4096 rows
  const size_t BTC = BT * kC;         // 4,194,304 elements

  const float* x   = (const float*)d_in[0];
  const float* Wq  = (const float*)d_in[1];
  const float* Wk  = (const float*)d_in[2];
  const float* Wv  = (const float*)d_in[3];
  const float* Wo  = (const float*)d_in[4];
  const float* bo  = (const float*)d_in[5];
  const float* W1  = (const float*)d_in[6];
  const float* b1  = (const float*)d_in[7];
  const float* W2  = (const float*)d_in[8];
  const float* b2  = (const float*)d_in[9];
  const float* g1  = (const float*)d_in[10];
  const float* be1 = (const float*)d_in[11];
  const float* g2  = (const float*)d_in[12];
  const float* be2 = (const float*)d_in[13];
  float* out = (float*)d_out;

  char* ws = (char*)d_ws;
  size_t off = 0;
  auto alloc = [&](size_t bytes) -> void* {
    void* p = ws + off;
    off += (bytes + 255) & ~(size_t)255;
    return p;
  };

  // Region A: x1h + q + k + vT (4 x 8 MiB) — reused as FFN hidden (32 MiB).
  _Float16* x1h = (_Float16*)alloc(BTC * 2);
  _Float16* qh  = (_Float16*)alloc(BTC * 2);
  _Float16* kh  = (_Float16*)alloc(BTC * 2);
  _Float16* vth = (_Float16*)alloc(BTC * 2);
  _Float16* hh  = x1h;  // alias: B*T x 4C halves == 32 MiB exactly
  _Float16* attnh = (_Float16*)alloc(BTC * 2);
  _Float16* x2h = attnh;  // alias: attn output consumed before x2 produced
  float* xmid = (float*)alloc(BTC * 4);
  _Float16* wqkvh = (_Float16*)alloc((size_t)kC * 3 * kH * kDH * 2);  // [C,3072]
  _Float16* woh = (_Float16*)alloc((size_t)kC * kC * 2);
  _Float16* w1h = (_Float16*)alloc((size_t)kC * 4 * kC * 2);
  _Float16* w2h = (_Float16*)alloc((size_t)4 * kC * kC * 2);

  // Weight prep
  pack_qkv<<<1024, 256, 0, stream>>>(Wq, wqkvh, 0);
  pack_qkv<<<1024, 256, 0, stream>>>(Wk, wqkvh, 1);
  pack_qkv<<<1024, 256, 0, stream>>>(Wv, wqkvh, 2);
  cvt_f32_f16<<<1024, 256, 0, stream>>>(Wo, woh, (size_t)kC * kC);
  cvt_f32_f16<<<2048, 256, 0, stream>>>(W1, w1h, (size_t)kC * 4 * kC);
  cvt_f32_f16<<<2048, 256, 0, stream>>>(W2, w2h, (size_t)4 * kC * kC);

  // 1) pre-LN (over T) -> f16
  ln_time_kernel<<<kB * (kC / 32), 256, 0, stream>>>(x, g1, be1, x1h);

  // 2) fused QKV projection: [4096,1024] x [1024,3072]
  gemm128<EP_QKV><<<dim3(3 * kH * kDH / 128, BT / 128), 256, 0, stream>>>(
      x1h, wqkvh, nullptr, nullptr, qh, kh, vth, 3 * kH * kDH, kC);

  // 3) causal attention (scale = C^-0.5, faithful to source)
  attn_kernel<<<dim3(kT / 64, kH, kB), 128, 0, stream>>>(qh, kh, vth, attnh,
                                                         1.0f / 32.0f);

  // 4) output projection + bias + residual(x) -> xmid (f32)
  gemm128<EP_F32><<<dim3(kC / 128, BT / 128), 256, 0, stream>>>(
      attnh, woh, bo, x, xmid, nullptr, nullptr, kC, kC);

  // 5) second pre-LN (over T) -> f16
  ln_time_kernel<<<kB * (kC / 32), 256, 0, stream>>>(xmid, g2, be2, x2h);

  // 6) FFN up + bias + ReLU -> f16 hidden
  gemm128<EP_F16R><<<dim3(4 * kC / 128, BT / 128), 256, 0, stream>>>(
      x2h, w1h, b1, nullptr, hh, nullptr, nullptr, 4 * kC, kC);

  // 7) FFN down + bias + residual(xmid) -> d_out (f32)
  gemm128<EP_F32><<<dim3(kC / 128, BT / 128), 256, 0, stream>>>(
      hh, w2h, b2, xmid, out, nullptr, nullptr, kC, 4 * kC);
}
